// StyleGan2GeneratorBlock_6768868458807
// MI455X (gfx1250) — matrix operand order repaired and checked by reading, not compiled, not run on hardware
//
#include <hip/hip_runtime.h>
#include <hip/hip_bf16.h>
#include <stdint.h>

// ---------------------------------------------------------------------------
// StyleGAN2 generator block for MI455X (gfx1250, wave32, WMMA bf16)
//
// Pipeline (all on `stream`, sequential kernels give ordering):
//   0. zero_kernel        : zero the two padded NHWC activation buffers
//   1. prep_weights (x2)  : W (fp32 OIHW) -> Wb bf16 [tap][O][Cin], sum W^2
//   2. styles_kernel      : s+1 for style1/style2/styleRGB   (8x512 each)
//   3. dcoef_kernel       : per-(b,o) demod coefficient 1/max(norm,1e-8)
//   4. upsample_mod       : x fp32 NCHW 32x32 -> bf16 NHWC 66x66 padded, *(s1+1)
//   5. conv_wmma<0>       : conv1 + noise + lrelu, writes x2pad bf16 NHWC *(s2+1)
//   6. conv_wmma<1>       : conv2 + noise + lrelu, writes d_out.x fp32 NCHW
//                           and xrgb bf16 NHWC *(srgb+1)
//   7. torgb_kernel       : 3x512 dot per pixel -> d_out.rgb
//
// Conv = implicit GEMM, per batch: M=4096 pixels, N=512 out-ch, K=9*512.
// K-loop is software-pipelined (register ping-pong) so fragment loads for
// step i+1 overlap the 16 WMMAs of step i.
// ---------------------------------------------------------------------------

typedef __attribute__((ext_vector_type(16))) __bf16 v16bf;
typedef __attribute__((ext_vector_type(8)))  float  v8f;

union FragAB {
  v16bf v;
  uint4 q[2];
};

// ---------------- constants ----------------
#define NB    8
#define NC    512
#define NH    64
#define NPIX  4096        // 64*64
#define NPAD  66          // 64 + 2 halo
#define PADPIX (NPAD*NPAD)

// ---------------- workspace layout (bytes) ----------------
static constexpr size_t SZ_S    = (size_t)NB * NC * sizeof(float);          // 16 KB
static constexpr size_t OFF_S1  = 0;
static constexpr size_t OFF_S2  = OFF_S1 + SZ_S;
static constexpr size_t OFF_SR  = OFF_S2 + SZ_S;
static constexpr size_t OFF_D1  = OFF_SR + SZ_S;
static constexpr size_t OFF_D2  = OFF_D1 + SZ_S;
static constexpr size_t SZ_SW   = (size_t)NC * NC * sizeof(float);          // 1 MB
static constexpr size_t OFF_SW1 = OFF_D2 + SZ_S;
static constexpr size_t OFF_SW2 = OFF_SW1 + SZ_SW;
static constexpr size_t SZ_WB   = (size_t)9 * NC * NC * 2;                  // 4.5 MB
static constexpr size_t OFF_WB1 = OFF_SW2 + SZ_SW;
static constexpr size_t OFF_WB2 = OFF_WB1 + SZ_WB;
static constexpr size_t SZ_XPAD = (size_t)NB * PADPIX * NC * 2;             // ~34 MB
static constexpr size_t OFF_X1  = OFF_WB2 + SZ_WB;
static constexpr size_t OFF_X2  = OFF_X1 + SZ_XPAD;
static constexpr size_t SZ_XR   = (size_t)NB * NPIX * NC * 2;               // 32 MB
static constexpr size_t OFF_XR  = OFF_X2 + SZ_XPAD;

// ---------------- 0: zero padded activation buffers ----------------
__global__ void zero_kernel(uint4* __restrict__ p, long n4) {
  long i = (long)blockIdx.x * blockDim.x + threadIdx.x;
  long stride = (long)gridDim.x * blockDim.x;
  uint4 z; z.x = 0u; z.y = 0u; z.z = 0u; z.w = 0u;
  for (; i < n4; i += stride) p[i] = z;
}

// ---------------- 1: weight prep: sum-of-squares + bf16 [tap][O][Cin] -------
__global__ void prep_weights(const float* __restrict__ W,
                             float* __restrict__ sumwsq,
                             __hip_bfloat16* __restrict__ wb) {
  int idx = blockIdx.x * 256 + threadIdx.x;     // o*512 + cin, 262144 total
  int o  = idx >> 9;
  int ci = idx & 511;
  const float* wp = W + (size_t)idx * 9;
  float s = 0.f;
#pragma unroll
  for (int t = 0; t < 9; ++t) {
    float w = wp[t];
    s += w * w;
    wb[((size_t)t * NC + o) * NC + ci] = __float2bfloat16(w);
  }
  sumwsq[idx] = s;
}

// ---------------- 2: styles (+1 folded in) ----------------
__global__ void styles_kernel(const float* __restrict__ wembs,
                              const float* __restrict__ Ws1, const float* __restrict__ bs1,
                              const float* __restrict__ Ws2, const float* __restrict__ bs2,
                              const float* __restrict__ Wsr, const float* __restrict__ bsr,
                              float* __restrict__ s1p, float* __restrict__ s2p,
                              float* __restrict__ srp) {
  int which = blockIdx.x >> 4;                  // 0,1,2
  int rem   = blockIdx.x & 15;
  int b     = rem >> 1;
  int o     = (rem & 1) * 256 + threadIdx.x;
  const float* Ws = (which == 0) ? Ws1 : (which == 1) ? Ws2 : Wsr;
  const float* bs = (which == 0) ? bs1 : (which == 1) ? bs2 : bsr;
  float*       ds = (which == 0) ? s1p : (which == 1) ? s2p : srp;
  const float* we = wembs + (size_t)b * NC;
  const float* wr = Ws + (size_t)o * NC;
  float acc = 0.f;
  for (int z = 0; z < NC; ++z) acc += we[z] * wr[z];
  ds[b * NC + o] = acc + bs[o] + 1.0f;
}

// ---------------- 3: demod coefficients ----------------
__global__ void dcoef_kernel(const float* __restrict__ s1p, const float* __restrict__ s2p,
                             const float* __restrict__ sw1, const float* __restrict__ sw2,
                             float* __restrict__ d1, float* __restrict__ d2) {
  int which = blockIdx.x >> 4;                  // 0,1
  int rem   = blockIdx.x & 15;
  int b     = rem >> 1;
  int o     = (rem & 1) * 256 + threadIdx.x;
  const float* sp = which ? s2p : s1p;
  const float* sw = which ? sw2 : sw1;
  float*       d  = which ? d2  : d1;
  const float* sb = sp + (size_t)b * NC;
  const float* wr = sw + (size_t)o * NC;
  float acc = 0.f;
  for (int c = 0; c < NC; ++c) {
    float s = sb[c];
    acc += s * s * wr[c];
  }
  float n = fmaxf(sqrtf(acc), 1e-8f);
  d[b * NC + o] = 1.0f / n;
}

// ---------------- 4: nearest-2x upsample + modulation -> padded NHWC bf16 ---
__global__ void upsample_mod(const float* __restrict__ x,
                             const float* __restrict__ s1p,
                             __hip_bfloat16* __restrict__ xmod) {
  int p = blockIdx.x & (NPIX - 1);
  int b = blockIdx.x >> 12;
  int y = p >> 6, xx = p & 63;
  int sy = y >> 1, sx = xx >> 1;
  const float* xb = x + ((size_t)b * NC) * 1024 + sy * 32 + sx;   // NCHW 32x32
  const float* sb = s1p + (size_t)b * NC;
  size_t obase = (size_t)b * PADPIX * NC + (size_t)((y + 1) * NPAD + (xx + 1)) * NC;
  for (int c = threadIdx.x; c < NC; c += 256) {
    float v = xb[(size_t)c * 1024] * sb[c];
    xmod[obase + c] = __float2bfloat16(v);
  }
}

// ---------------- 5/6: implicit-GEMM 3x3 conv via WMMA bf16 ----------------
// A (16x32 bf16) per lane: row M=ln, K = {8h..8h+7} u {16+8h..23+8h}
// B (32x16 bf16) per lane: col N=ln, K = {8h..8h+7} u {16+8h..23+8h}
// C (16x16 f32)  per lane: element (M = r+8h, N = ln)
__device__ __forceinline__ void load_frags(FragAB (&A)[4], FragAB (&B)[4],
                                           const __hip_bfloat16* const (&abase)[4],
                                           const __hip_bfloat16* const (&bbase)[4],
                                           int c0) {
#pragma unroll
  for (int f = 0; f < 4; ++f) {
    A[f].q[0] = *(const uint4*)(abase[f] + c0);        // K 8h..8h+7
    A[f].q[1] = *(const uint4*)(abase[f] + c0 + 16);   // K 16+8h..23+8h
  }
#pragma unroll
  for (int g = 0; g < 4; ++g) {
    B[g].q[0] = *(const uint4*)(bbase[g] + c0);        // K 8h..8h+7
    B[g].q[1] = *(const uint4*)(bbase[g] + c0 + 16);   // K 16+8h..23+8h
  }
}

template <int MODE>
__global__ __launch_bounds__(256) void conv_wmma_kernel(
    const __hip_bfloat16* __restrict__ xin,    // padded NHWC [B][66][66][512]
    const __hip_bfloat16* __restrict__ wb,     // [9][O][Cin] bf16
    const float* __restrict__ dcoef,           // [B][512]
    const float* __restrict__ bias,            // [512]
    const float* __restrict__ noise,           // [B][4096]
    const float* __restrict__ nscale,          // [512]
    const float* __restrict__ smod,            // [B][512]  next-layer style (+1)
    __hip_bfloat16* __restrict__ out_bf,       // MODE0: x2pad ; MODE1: xrgb
    float* __restrict__ out_f32)               // MODE1: d_out x-part (NCHW)
{
  const int lane = threadIdx.x & 31;
  const int wave = threadIdx.x >> 5;
  const int wm = wave & 1;                     // 2 pixel-waves
  const int wn = wave >> 1;                    // 4 channel-waves
  const int b  = blockIdx.z;
  const int pix0 = blockIdx.x * 128 + wm * 64;
  const int och0 = blockIdx.y * 256 + wn * 64;
  const int h  = lane >> 4;
  const int ln = lane & 15;

  int py[4], px[4];
#pragma unroll
  for (int f = 0; f < 4; ++f) {
    int p = pix0 + f * 16 + ln;
    py[f] = p >> 6;
    px[f] = p & 63;
  }
  const __hip_bfloat16* xb = xin + (size_t)b * PADPIX * NC;

  v8f zero = {};
  v8f acc[4][4];
#pragma unroll
  for (int f = 0; f < 4; ++f)
#pragma unroll
    for (int g = 0; g < 4; ++g) acc[f][g] = zero;

  for (int t = 0; t < 9; ++t) {
    const int dy = t / 3, dx = t % 3;
    const __hip_bfloat16* abase[4];
#pragma unroll
    for (int f = 0; f < 4; ++f)
      abase[f] = xb + ((size_t)((py[f] + dy) * NPAD + (px[f] + dx)) * NC + 8 * h);
    const __hip_bfloat16* bbase[4];
#pragma unroll
    for (int g = 0; g < 4; ++g)
      bbase[g] = wb + ((size_t)(t * NC + och0 + g * 16 + ln) * NC + 8 * h);

    // Software pipeline: prefetch K-step i+1 while WMMAs run on step i.
    FragAB A[2][4], B[2][4];
    load_frags(A[0], B[0], abase, bbase, 0);
#pragma unroll 1
    for (int c0 = 0; c0 < NC; c0 += 32) {
      const int cur = (c0 >> 5) & 1;
      const int nxt = cur ^ 1;
      if (c0 + 32 < NC)
        load_frags(A[nxt], B[nxt], abase, bbase, c0 + 32);
#pragma unroll
      for (int f = 0; f < 4; ++f)
#pragma unroll
        for (int g = 0; g < 4; ++g)
          acc[f][g] = __builtin_amdgcn_wmma_f32_16x16x32_bf16(
              false, A[cur][f].v, false, B[cur][g].v, (short)0, acc[f][g],
              false, false);
    }
  }

  // -------- epilogue: demod, bias, noise, leaky-relu, fused next-mod -------
  const float* dcb = dcoef  + (size_t)b * NC;
  const float* smb = smod   + (size_t)b * NC;
  const float* nzb = noise  + (size_t)b * NPIX;

  int   oarr[4];
  float dc[4], bi[4], sc[4], sm[4];
#pragma unroll
  for (int g = 0; g < 4; ++g) {
    const int o = och0 + g * 16 + ln;
    oarr[g] = o;
    dc[g] = dcb[o];
    bi[g] = bias[o];
    sc[g] = nscale[o];
    sm[g] = smb[o];
  }

#pragma unroll
  for (int f = 0; f < 4; ++f) {
#pragma unroll
    for (int r = 0; r < 8; ++r) {
      const int p = pix0 + f * 16 + r + 8 * h;
      const float nz = nzb[p];
      const int yy = (p >> 6) + 1, xx = (p & 63) + 1;
      const size_t pad_base = (size_t)b * PADPIX * NC + (size_t)(yy * NPAD + xx) * NC;
      const size_t rgb_base = ((size_t)b * NPIX + p) * NC;
#pragma unroll
      for (int g = 0; g < 4; ++g) {
        float val = acc[f][g][r] * dc[g] + bi[g] + nz * sc[g];
        val = (val > 0.0f) ? val : 0.2f * val;
        if (MODE == 0) {
          out_bf[pad_base + oarr[g]] = __float2bfloat16(val * sm[g]);
        } else {
          out_f32[((size_t)b * NC + oarr[g]) * NPIX + p] = val;
          out_bf[rgb_base + oarr[g]] = __float2bfloat16(val * sm[g]);
        }
      }
    }
  }
}

// ---------------- 7: ToRGB (1x1 conv, style pre-folded into xrgb) ----------
__global__ void torgb_kernel(const __hip_bfloat16* __restrict__ xrgb,
                             const float* __restrict__ Wrgb,
                             const float* __restrict__ brgb,
                             float* __restrict__ out) {
  int p = blockIdx.x * 256 + threadIdx.x;       // 32768 (b,pixel) pairs
  int b = p >> 12;
  int pix = p & (NPIX - 1);
  const __hip_bfloat16* xr = xrgb + (size_t)p * NC;
  float a0 = 0.f, a1 = 0.f, a2 = 0.f;
  for (int c = 0; c < NC; ++c) {
    float v = __bfloat162float(xr[c]);
    a0 += v * Wrgb[c];
    a1 += v * Wrgb[NC + c];
    a2 += v * Wrgb[2 * NC + c];
  }
  size_t base = (size_t)NB * NC * NPIX;
  out[base + ((size_t)b * 3 + 0) * NPIX + pix] = a0 + brgb[0];
  out[base + ((size_t)b * 3 + 1) * NPIX + pix] = a1 + brgb[1];
  out[base + ((size_t)b * 3 + 2) * NPIX + pix] = a2 + brgb[2];
}

// ---------------------------------------------------------------------------
extern "C" void kernel_launch(void* const* d_in, const int* in_sizes, int n_in,
                              void* d_out, int out_size, void* d_ws, size_t ws_size,
                              hipStream_t stream) {
  const float* x      = (const float*)d_in[0];
  const float* wembs  = (const float*)d_in[1];
  const float* W1     = (const float*)d_in[2];
  const float* b1     = (const float*)d_in[3];
  const float* Ws1    = (const float*)d_in[4];
  const float* bs1    = (const float*)d_in[5];
  const float* scale1 = (const float*)d_in[6];
  const float* noise1 = (const float*)d_in[7];
  const float* W2     = (const float*)d_in[8];
  const float* b2     = (const float*)d_in[9];
  const float* Ws2    = (const float*)d_in[10];
  const float* bs2    = (const float*)d_in[11];
  const float* scale2 = (const float*)d_in[12];
  const float* noise2 = (const float*)d_in[13];
  const float* Wrgb   = (const float*)d_in[14];
  const float* brgb   = (const float*)d_in[15];
  const float* Wsrgb  = (const float*)d_in[16];
  const float* bsrgb  = (const float*)d_in[17];

  char* ws = (char*)d_ws;
  float* s1p = (float*)(ws + OFF_S1);
  float* s2p = (float*)(ws + OFF_S2);
  float* srp = (float*)(ws + OFF_SR);
  float* d1  = (float*)(ws + OFF_D1);
  float* d2  = (float*)(ws + OFF_D2);
  float* sw1 = (float*)(ws + OFF_SW1);
  float* sw2 = (float*)(ws + OFF_SW2);
  __hip_bfloat16* wb1  = (__hip_bfloat16*)(ws + OFF_WB1);
  __hip_bfloat16* wb2  = (__hip_bfloat16*)(ws + OFF_WB2);
  __hip_bfloat16* x1p  = (__hip_bfloat16*)(ws + OFF_X1);
  __hip_bfloat16* x2p  = (__hip_bfloat16*)(ws + OFF_X2);
  __hip_bfloat16* xrgb = (__hip_bfloat16*)(ws + OFF_XR);
  float* out = (float*)d_out;

  // 0: zero both padded activation buffers (x1p, x2p are contiguous)
  {
    long n4 = (long)(2 * SZ_XPAD / 16);
    zero_kernel<<<1024, 256, 0, stream>>>((uint4*)x1p, n4);
  }
  // 1: weight prep (262144 threads each)
  prep_weights<<<1024, 256, 0, stream>>>(W1, sw1, wb1);
  prep_weights<<<1024, 256, 0, stream>>>(W2, sw2, wb2);
  // 2: styles (+1)
  styles_kernel<<<48, 256, 0, stream>>>(wembs, Ws1, bs1, Ws2, bs2, Wsrgb, bsrgb,
                                        s1p, s2p, srp);
  // 3: demod coefficients
  dcoef_kernel<<<32, 256, 0, stream>>>(s1p, s2p, sw1, sw2, d1, d2);
  // 4: upsample + modulate into padded NHWC bf16
  upsample_mod<<<NB * NPIX, 256, 0, stream>>>(x, s1p, x1p);
  // 5: conv1 (fuses noise, lrelu, *(style2+1))
  {
    dim3 grid(NPIX / 128, NC / 256, NB);
    conv_wmma_kernel<0><<<grid, 256, 0, stream>>>(x1p, wb1, d1, b1, noise1,
                                                  scale1, s2p, x2p, nullptr);
  }
  // 6: conv2 (fuses noise, lrelu; writes fp32 x output + *(styleRGB+1) bf16)
  {
    dim3 grid(NPIX / 128, NC / 256, NB);
    conv_wmma_kernel<1><<<grid, 256, 0, stream>>>(x2p, wb2, d2, b2, noise2,
                                                  scale2, srp, xrgb, out);
  }
  // 7: ToRGB
  torgb_kernel<<<NB * NPIX / 256, 256, 0, stream>>>(xrgb, Wrgb, brgb, out);
}
